// SimpleLSTM_27401891348775
// MI455X (gfx1250) — matrix operand order repaired and checked by reading, not compiled, not run on hardware
//
#include <hip/hip_runtime.h>

typedef __bf16 bf16;
typedef __attribute__((ext_vector_type(16))) __bf16 v16bf;
typedef __attribute__((ext_vector_type(8)))  __bf16 v8bf;
typedef __attribute__((ext_vector_type(8)))  float  v8f;

#define B_  64
#define T_  512
#define D_  512
#define U_  1024
#define NG_ 4096       // 4*U
#define K_  1536       // D + U
#define NBLOCKS_ 64    // U / 16
#define NTHREADS_ 512  // 16 waves: one 16x16 tile each (4 gates x 4 m-tiles)

// ---- workspace layout (bytes) ----
#define XBF_BYTES 33554432ull   // Xbf [T,B,D] bf16
#define WT_BYTES  12582912ull   // WT  [NG,K]  bf16 (transposed weight concat)
#define H_BYTES   131072ull     // each h ping-pong buffer [B,U] bf16

__device__ __forceinline__ float fsig(float x) { return 1.0f / (1.0f + __expf(-x)); }
__device__ __forceinline__ float ftanh(float x) { return 2.0f / (1.0f + __expf(-2.0f * x)) - 1.0f; }

// x [B,T,D] f32  ->  Xbf [T,B,D] bf16
__global__ void conv_x_kernel(const float* __restrict__ x, bf16* __restrict__ Xbf) {
    size_t o = (size_t)blockIdx.x * blockDim.x + threadIdx.x;   // flat over T*B*D
    int d = (int)(o & (D_ - 1));
    size_t r = o >> 9;
    int b = (int)(r & (B_ - 1));
    int t = (int)(r >> 6);
    Xbf[o] = (bf16)x[((size_t)b * T_ + t) * D_ + d];
}

// WT [NG,K] bf16 from kernel [D,NG] and rec_kernel [U,NG] (transposed concat)
__global__ void conv_w_kernel(const float* __restrict__ w, const float* __restrict__ r,
                              bf16* __restrict__ WT) {
    size_t o = (size_t)blockIdx.x * blockDim.x + threadIdx.x;   // flat over NG*K
    int k = (int)(o % K_);
    int n = (int)(o / K_);
    float v = (k < D_) ? w[(size_t)k * NG_ + n] : r[(size_t)(k - D_) * NG_ + n];
    WT[o] = (bf16)v;
}

// zero h ping-pong buffers (contiguous H0|H1) and barrier words
__global__ void init_state_kernel(bf16* __restrict__ H, unsigned* __restrict__ bar) {
    size_t o = (size_t)blockIdx.x * blockDim.x + threadIdx.x;
    if (o < 2ull * B_ * U_) H[o] = (bf16)0.0f;
    if (o < 32) bar[o] = 0u;
}

__global__ __launch_bounds__(NTHREADS_, 1)
void lstm_persistent_kernel(const bf16* __restrict__ Xbf,
                            const bf16* __restrict__ WT,
                            bf16* __restrict__ H0,
                            bf16* __restrict__ H1,
                            unsigned* __restrict__ bar,
                            float* __restrict__ out) {
    // LDS-resident weight slice for this block's 64 output columns (4 gates x 16 units):
    // blds[cidx][k], cidx = gate*16 + unit, k = 0..1535.  192 KB, loaded once, reused 512 steps.
    __shared__ bf16  blds[64 * K_];     // 196,608 B
    __shared__ float zlds[4][B_][16];   //  16 KB: z for 4 gates x 64 batch x 16 units
    __shared__ float clds[B_][16];      //   4 KB: persistent cell state

    const int tid  = threadIdx.x;
    const int lane = tid & 31;
    const int wave = tid >> 5;              // 0..15
    const int gate = wave & 3;              // gate strip (i,f,g,o)
    const int mrow = (wave >> 2) * 16;      // batch-row tile base: 0/16/32/48
    const int u0   = blockIdx.x * 16;       // unit slice owned by this block

    // WMMA lane mapping (per ISA 7.12.2):
    const int nloc  = lane & 15;            // B/C/D column within tile
    const int kb_a  = (lane >> 4) * 8;      // A: lanes 16-31 hold K+8 block
    const int kb_b  = (lane >> 4) * 8;      // B: lanes 16-31 hold K+8 block
    const int mbase = (lane >> 4) * 8;      // C/D: lanes 16-31 hold M+8
    const int arow  = mrow + nloc;          // A row (lane = row within tile)

    // ---- preload weight slice into LDS (once) + init cell state ----
    for (int e = tid; e < B_ * 16; e += NTHREADS_) clds[e >> 4][e & 15] = 0.0f;
    for (int j = tid; j < 64 * (K_ / 8); j += NTHREADS_) {
        int cidx = j / (K_ / 8);
        int koff = (j % (K_ / 8)) * 8;
        int g = cidx >> 4, n = cidx & 15;
        const bf16* src = WT + ((size_t)g * U_ + u0 + n) * K_ + koff;
        *(v8bf*)&blds[cidx * K_ + koff] = *(const v8bf*)src;
    }
    __syncthreads();

    const bf16* bbase = &blds[(gate * 16 + nloc) * K_];   // this lane's B column
    unsigned* cnt = bar;
    unsigned* gen = bar + 16;

    for (int t = 0; t < T_; ++t) {
        const bf16* Hread  = (t & 1) ? H1 : H0;
        bf16*       Hwrite = (t & 1) ? H0 : H1;

        v8f acc = {0.f,0.f,0.f,0.f,0.f,0.f,0.f,0.f};

        const bf16* Xr = Xbf + ((size_t)t * B_ + arow) * D_;
        if (t + 1 < T_) __builtin_prefetch(Xr + (size_t)B_ * D_, 0, 0);

        // ---- K part 1: input contribution (A = x_t, K = 0..511), B from LDS ----
        #pragma unroll 4
        for (int kc = 0; kc < D_; kc += 32) {
            v16bf bfrag = __builtin_shufflevector(*(const v8bf*)(bbase + kc + kb_b), *(const v8bf*)(bbase + kc + kb_b + 16), 0,1,2,3,4,5,6,7,8,9,10,11,12,13,14,15);
            v8bf alo = *(const v8bf*)(Xr + kc + kb_a);
            v8bf ahi = *(const v8bf*)(Xr + kc + kb_a + 16);
            v16bf a = __builtin_shufflevector(alo, ahi, 0,1,2,3,4,5,6,7,8,9,10,11,12,13,14,15);
            acc = __builtin_amdgcn_wmma_f32_16x16x32_bf16(false, a, false, bfrag, (short)0, acc, false, false);
        }

        // ---- K part 2: recurrent contribution (A = h_t, K = 512..1535) ----
        const bf16* Hr = Hread + (size_t)arow * U_;
        const bf16* bbH = bbase + D_;
        #pragma unroll 4
        for (int kc = 0; kc < U_; kc += 32) {
            v16bf bfrag = __builtin_shufflevector(*(const v8bf*)(bbH + kc + kb_b), *(const v8bf*)(bbH + kc + kb_b + 16), 0,1,2,3,4,5,6,7,8,9,10,11,12,13,14,15);
            v8bf alo = *(const v8bf*)(Hr + kc + kb_a);
            v8bf ahi = *(const v8bf*)(Hr + kc + kb_a + 16);
            v16bf a = __builtin_shufflevector(alo, ahi, 0,1,2,3,4,5,6,7,8,9,10,11,12,13,14,15);
            acc = __builtin_amdgcn_wmma_f32_16x16x32_bf16(false, a, false, bfrag, (short)0, acc, false, false);
        }

        // ---- spill z tile to LDS (D layout: vgpr v -> M = mbase+v) ----
        #pragma unroll
        for (int v = 0; v < 8; ++v)
            zlds[gate][mrow + mbase + v][nloc] = acc[v];
        __syncthreads();

        // ---- LSTM cell update (block-local: 64 batches x 16 units) ----
        #pragma unroll
        for (int i = 0; i < 2; ++i) {
            int e = tid + i * NTHREADS_;
            int b = e >> 4, n = e & 15;
            float zi = zlds[0][b][n];
            float zf = zlds[1][b][n];
            float zg = zlds[2][b][n];
            float zo = zlds[3][b][n];
            float cn = fsig(zf) * clds[b][n] + fsig(zi) * ftanh(zg);
            clds[b][n] = cn;
            float h = fsig(zo) * ftanh(cn);
            Hwrite[(size_t)b * U_ + u0 + n] = (bf16)h;
            if (t == T_ - 1) out[(size_t)b * U_ + u0 + n] = h;
        }

        // ---- device-wide barrier (double-buffered h => one barrier/step) ----
        __threadfence();
        __syncthreads();
        if (tid == 0) {
            unsigned g0 = __hip_atomic_load(gen, __ATOMIC_ACQUIRE, __HIP_MEMORY_SCOPE_AGENT);
            if (atomicAdd(cnt, 1u) == NBLOCKS_ - 1) {
                atomicExch(cnt, 0u);
                __hip_atomic_fetch_add(gen, 1u, __ATOMIC_ACQ_REL, __HIP_MEMORY_SCOPE_AGENT);
            } else {
                while (__hip_atomic_load(gen, __ATOMIC_ACQUIRE, __HIP_MEMORY_SCOPE_AGENT) == g0) {
                    __builtin_amdgcn_s_sleep(2);
                }
            }
        }
        __syncthreads();
    }
}

extern "C" void kernel_launch(void* const* d_in, const int* in_sizes, int n_in,
                              void* d_out, int out_size, void* d_ws, size_t ws_size,
                              hipStream_t stream) {
    (void)in_sizes; (void)n_in; (void)out_size; (void)ws_size;
    const float* x = (const float*)d_in[0];        // [B, T, D] f32
    const float* w = (const float*)d_in[1];        // [D, 4U]  f32
    const float* r = (const float*)d_in[2];        // [U, 4U]  f32
    float* out = (float*)d_out;                    // [B, U]   f32

    char* ws = (char*)d_ws;
    bf16* Xbf = (bf16*)ws;
    bf16* WT  = (bf16*)(ws + XBF_BYTES);
    bf16* H0  = (bf16*)(ws + XBF_BYTES + WT_BYTES);
    bf16* H1  = H0 + (size_t)B_ * U_;
    unsigned* bar = (unsigned*)(ws + XBF_BYTES + WT_BYTES + 2 * H_BYTES);

    conv_x_kernel<<<(B_ * T_ * D_) / 256, 256, 0, stream>>>(x, Xbf);
    conv_w_kernel<<<((size_t)NG_ * K_) / 256, 256, 0, stream>>>(w, r, WT);
    init_state_kernel<<<(2 * B_ * U_ + 255) / 256, 256, 0, stream>>>(H0, bar);
    lstm_persistent_kernel<<<NBLOCKS_, NTHREADS_, 0, stream>>>(Xbf, WT, H0, H1, bar, out);
}
